// ConcatenatedLoRALinearSidecarLayer_22153441312800
// MI455X (gfx1250) — matrix operand order repaired and checked by reading, not compiled, not run on hardware
//
#include <hip/hip_runtime.h>
#include <hip/hip_bf16.h>

#define B_   4
#define S_   4096
#define D_   4096
#define NL   3
#define R_   128
#define O_   4096
#define M_   (B_ * S_)
#define LSTR 40   // LDS row stride in bf16: 80B -> 16B-aligned chunks, conflict-free

typedef __bf16 bf16;
typedef __attribute__((ext_vector_type(8)))  __bf16 v8bf;
typedef __attribute__((ext_vector_type(16))) __bf16 v16bf;
typedef __attribute__((ext_vector_type(8)))  float  v8f;
typedef __attribute__((ext_vector_type(4)))  int    v4i;

#if defined(__has_builtin)
#if __has_builtin(__builtin_amdgcn_global_load_async_to_lds_b128) && \
    __has_builtin(__builtin_amdgcn_s_wait_asynccnt)
#define HAS_ASYNC 1
#endif
#endif
#ifndef HAS_ASYNC
#define HAS_ASYNC 0
#endif

// Native RNE f32->bf16 (vectorized; lowers to packed cvt on gfx1250)
static __device__ __forceinline__ v8bf cvt8(v8f f) {
    return __builtin_convertvector(f, v8bf);
}
static __device__ __forceinline__ bf16 f2bf(float f) {   // scalar (prep only)
    unsigned u = __builtin_bit_cast(unsigned, f);
    u += 0x7FFFu + ((u >> 16) & 1u);
    unsigned short s = (unsigned short)(u >> 16);
    return __builtin_bit_cast(bf16, s);
}

// 32B global -> LDS tile copy: async-to-LDS when available, else via VGPRs.
static __device__ __forceinline__ void cp32_to_lds(const bf16* g, bf16* l) {
#if HAS_ASYNC
    __builtin_amdgcn_global_load_async_to_lds_b128((v4i*)g, (v4i*)l, 0, 0);
    __builtin_amdgcn_global_load_async_to_lds_b128((v4i*)(g + 8), (v4i*)(l + 8), 0, 0);
#else
    *(v8bf*)l       = *(const v8bf*)g;
    *(v8bf*)(l + 8) = *(const v8bf*)(g + 8);
#endif
}
static __device__ __forceinline__ void async_fence() {
#if HAS_ASYNC
    __builtin_amdgcn_s_wait_asynccnt(0);
#endif
}

// Row-major LDS tile fragment gather: each lane reads two 16B chunks from its row.
static __device__ __forceinline__ v16bf frag_ld(const bf16* tile, int rowBase, int lane,
                                                int c0, int c1) {
    const bf16* p = tile + (size_t)(rowBase + (lane & 15)) * LSTR;
    v8bf lo = *(const v8bf*)(p + c0);
    v8bf hi = *(const v8bf*)(p + c1);
    v16bf r;
#pragma unroll
    for (int i = 0; i < 8; ++i) { r[i] = lo[i]; r[i + 8] = hi[i]; }
    return r;
}
// A fragment (16x32): lane<16 holds K 0-7 & 16-23; lane>=16 holds K 8-15 & 24-31
static __device__ __forceinline__ v16bf fragA(const bf16* tile, int rowBase, int lane) {
    const int half = lane >> 4;
    return frag_ld(tile, rowBase, lane, half * 8, 16 + half * 8);
}
// B fragment (32x16, stored [col][k]): lane<16 col j K 0-7 & 16-23; lane>=16 K 8-15 & 24-31
static __device__ __forceinline__ v16bf fragB(const bf16* tile, int colBase, int lane) {
    const int half = lane >> 4;
    return frag_ld(tile, colBase, lane, half * 8, 16 + half * 8);
}

// ---------------------------------------------------------------- prep: cvt + fold scales
__global__ __launch_bounds__(256) void lora_prep(
    const float* __restrict__ down, const float* __restrict__ up,
    const float* __restrict__ bias, const float* __restrict__ scales,
    bf16* __restrict__ down_b, bf16* __restrict__ up_b, float* __restrict__ bias_s,
    unsigned nd, unsigned nu, unsigned nb) {
    unsigned i = blockIdx.x * blockDim.x + threadIdx.x;
    if (i < nd) {
        down_b[i] = f2bf(down[i]);
    } else if (i < nd + nu) {
        unsigned j = i - nd;
        unsigned n = j / (O_ * R_);
        up_b[j] = f2bf(up[j] * (0.8f * scales[n]));
    } else if (i < nd + nu + nb) {
        unsigned j = i - nd - nu;
        unsigned n = j / O_;
        bias_s[j] = bias[j] * (0.8f * scales[n]);
    }
}

// ---------------------------------------------------------------- stage 1: h = X @ down^T
__global__ __launch_bounds__(256) void lora_stage1(
    const float* __restrict__ x, const bf16* __restrict__ dwn, bf16* __restrict__ h) {
    __shared__ __attribute__((aligned(16))) bf16 xs[128 * LSTR];
    __shared__ __attribute__((aligned(16))) bf16 dsb[128 * LSTR];
    const int tid  = threadIdx.x;
    const int wave = tid >> 5, lane = tid & 31;
    const int wr = (wave >> 1) * 32;      // wave row base within 128-row tile
    const int wc = (wave & 1) * 64;       // wave col base within 128-col tile (== R)
    const int row0 = blockIdx.x * 128;
    const int n    = blockIdx.y;
    const int lrow = tid >> 1, lkc = (tid & 1) * 16;   // coop-load mapping
    const float* xrow = x   + (size_t)(row0 + lrow) * D_;
    const bf16*  drow = dwn + ((size_t)n * R_ + lrow) * D_;
    bf16* xdst = xs  + lrow * LSTR + lkc;
    bf16* ddst = dsb + lrow * LSTR + lkc;

    v8f acc[2][4] = {};
    for (int kt = 0; kt < D_ / 32; ++kt) {
        const int kb = kt * 32 + lkc;
        // down (pre-converted bf16): 32B copy straight into LDS (async if available)
        cp32_to_lds(drow + kb, ddst);
        // x: 16 f32 -> 16 bf16 -> LDS
        float4 f0 = *(const float4*)(xrow + kb + 0);
        float4 f1 = *(const float4*)(xrow + kb + 4);
        float4 f2 = *(const float4*)(xrow + kb + 8);
        float4 f3 = *(const float4*)(xrow + kb + 12);
        v8f u0 = {f0.x, f0.y, f0.z, f0.w, f1.x, f1.y, f1.z, f1.w};
        v8f u1 = {f2.x, f2.y, f2.z, f2.w, f3.x, f3.y, f3.z, f3.w};
        *(v8bf*)(xdst)     = cvt8(u0);
        *(v8bf*)(xdst + 8) = cvt8(u1);
        async_fence();
        __syncthreads();
        v16bf a[2], b[4];
#pragma unroll
        for (int tm = 0; tm < 2; ++tm) a[tm] = fragA(xs, wr + tm * 16, lane);
#pragma unroll
        for (int tn = 0; tn < 4; ++tn) b[tn] = fragB(dsb, wc + tn * 16, lane);
#pragma unroll
        for (int tm = 0; tm < 2; ++tm)
#pragma unroll
            for (int tn = 0; tn < 4; ++tn)
                acc[tm][tn] = __builtin_amdgcn_wmma_f32_16x16x32_bf16(
                    false, a[tm], false, b[tn], (short)0, acc[tm][tn], false, false);
        __syncthreads();
    }
    // store h as bf16: lane<16 -> col=lane, rows half*8..+7 of the 16x16 tile
    const int half = lane >> 4, nn = lane & 15;
    bf16* hb = h + (size_t)n * M_ * R_;
#pragma unroll
    for (int tm = 0; tm < 2; ++tm)
#pragma unroll
        for (int tn = 0; tn < 4; ++tn) {
            const int rcol = wc + tn * 16 + nn;
            v8bf hv = cvt8(acc[tm][tn]);
#pragma unroll
            for (int i = 0; i < 8; ++i) {
                const int mrow = row0 + wr + tm * 16 + half * 8 + i;
                hb[(size_t)mrow * R_ + rcol] = hv[i];
            }
        }
}

// ---------------------------------------------------------------- stage 2: y = h @ up^T + bias
__global__ __launch_bounds__(256) void lora_stage2(
    const bf16* __restrict__ h, const bf16* __restrict__ up,
    const float* __restrict__ bias_s, float* __restrict__ out) {
    __shared__ __attribute__((aligned(16))) bf16 hsb[128 * LSTR];
    __shared__ __attribute__((aligned(16))) bf16 usb[128 * LSTR];
    const int tid  = threadIdx.x;
    const int wave = tid >> 5, lane = tid & 31;
    const int wr = (wave >> 1) * 32, wc = (wave & 1) * 64;
    const int row0 = blockIdx.x * 128;
    const int n    = blockIdx.y >> 5;           // 32 col-blocks per branch (O/128)
    const int ob   = (blockIdx.y & 31) * 128;
    const int lrow = tid >> 1, lkc = (tid & 1) * 16;
    const bf16* hrow = h  + ((size_t)n * M_ + row0 + lrow) * R_;
    const bf16* urow = up + ((size_t)n * O_ + ob + lrow) * R_;
    bf16* hdst = hsb + lrow * LSTR + lkc;
    bf16* udst = usb + lrow * LSTR + lkc;

    v8f acc[2][4] = {};
#pragma unroll
    for (int kt = 0; kt < R_ / 32; ++kt) {
        const int kb = kt * 32 + lkc;
        cp32_to_lds(hrow + kb, hdst);
        cp32_to_lds(urow + kb, udst);
        async_fence();
        __syncthreads();
        v16bf a[2], b[4];
#pragma unroll
        for (int tm = 0; tm < 2; ++tm) a[tm] = fragA(hsb, wr + tm * 16, lane);
#pragma unroll
        for (int tn = 0; tn < 4; ++tn) b[tn] = fragB(usb, wc + tn * 16, lane);
#pragma unroll
        for (int tm = 0; tm < 2; ++tm)
#pragma unroll
            for (int tn = 0; tn < 4; ++tn)
                acc[tm][tn] = __builtin_amdgcn_wmma_f32_16x16x32_bf16(
                    false, a[tm], false, b[tn], (short)0, acc[tm][tn], false, false);
        __syncthreads();
    }
    const int half = lane >> 4, nn = lane & 15;
#pragma unroll
    for (int tm = 0; tm < 2; ++tm)
#pragma unroll
        for (int tn = 0; tn < 4; ++tn) {
            const int colL = wc + tn * 16 + nn;
            const float bv = bias_s[(size_t)n * O_ + ob + colL];
#pragma unroll
            for (int i = 0; i < 8; ++i) {
                const int mrow = row0 + wr + tm * 16 + half * 8 + i;
                out[(size_t)mrow * (NL * O_) + (size_t)n * O_ + ob + colL] =
                    acc[tm][tn][i] + bv;
            }
        }
}

extern "C" void kernel_launch(void* const* d_in, const int* in_sizes, int n_in,
                              void* d_out, int out_size, void* d_ws, size_t ws_size,
                              hipStream_t stream) {
    (void)in_sizes; (void)n_in; (void)out_size; (void)ws_size;
    const float* x      = (const float*)d_in[0];
    const float* down   = (const float*)d_in[1];
    const float* up     = (const float*)d_in[2];
    const float* bias   = (const float*)d_in[3];
    const float* scales = (const float*)d_in[4];
    float* out = (float*)d_out;

    const size_t nd = (size_t)NL * R_ * D_;   // 1,572,864
    const size_t nu = (size_t)NL * O_ * R_;   // 1,572,864
    const size_t nb = (size_t)NL * O_;        // 12,288
    char* w = (char*)d_ws;                    // all offsets 256B-aligned
    bf16*  down_b = (bf16*)(w);
    bf16*  up_b   = (bf16*)(w + nd * 2);
    float* bias_s = (float*)(w + nd * 2 + nu * 2);
    bf16*  h      = (bf16*)(w + nd * 2 + nu * 2 + nb * 4);   // 12 MB bf16

    const unsigned tot = (unsigned)(nd + nu + nb);
    lora_prep<<<(tot + 255) / 256, 256, 0, stream>>>(
        down, up, bias, scales, down_b, up_b, bias_s,
        (unsigned)nd, (unsigned)nu, (unsigned)nb);
    lora_stage1<<<dim3(M_ / 128, NL), 256, 0, stream>>>(x, down_b, h);
    lora_stage2<<<dim3(M_ / 128, NL * (O_ / 128)), 256, 0, stream>>>(h, up_b, bias_s, out);
}
